// VanillaRNN_30133490549111
// MI455X (gfx1250) — matrix operand order repaired and checked by reading, not compiled, not run on hardware
//
#include <hip/hip_runtime.h>
#include <stdint.h>

typedef __attribute__((ext_vector_type(16))) _Float16 v16h;
typedef __attribute__((ext_vector_type(8)))  float    v8f;
typedef __attribute__((ext_vector_type(2)))  _Float16 h2v;
typedef __attribute__((ext_vector_type(2)))  float    f2v;

#define Bsz 8192
#define Ssz 128
#define Hsz 512
#define Csz 256
#define BT  64

__device__ __forceinline__ v8f wmma_f16(v16h a, v16h b, v8f c) {
  // (neg_a, A, neg_b, B, c_mod, C, reuse_a, reuse_b)
  return __builtin_amdgcn_wmma_f32_16x16x32_f16(false, a, false, b, (short)0, c,
                                                false, false);
}

__device__ __forceinline__ float fast_tanh(float x) {
#if __has_builtin(__builtin_amdgcn_tanhf)
  return __builtin_amdgcn_tanhf(x);   // v_tanh_f32 (gfx1250 trans op)
#else
  return tanhf(x);
#endif
}

// ---------------------------------------------------------------------------
// Prep 1: W_hh (512x256), W_ph (256x512), W_hx (512x128) f32 -> f16 in WMMA
// A-fragment order [mtile][kchunk][lane][j]. k order used here (same in B): lane holds
// row M = lane&15; element j -> K = j + 16*(lane>>4) in the chunk.
// ---------------------------------------------------------------------------
__global__ __launch_bounds__(256)
void prep_weights(const float* __restrict__ Whh, const float* __restrict__ Wph,
                  const float* __restrict__ Whx,
                  _Float16* __restrict__ whhF, _Float16* __restrict__ wphF,
                  _Float16* __restrict__ whxF) {
  int i = blockIdx.x * blockDim.x + threadIdx.x;   // 0 .. 327679
  int f, j, lane, kf, kh, KA;
  if (i < 131072) {                                // W_hh: 32 Mtiles x 8 K
    f = i; j = f & 15; lane = (f >> 4) & 31; kf = f >> 9;
    kh = lane >> 4; KA = j + 16 * kh;
    int mt = kf >> 3, k = kf & 7;
    whhF[f] = (_Float16)Whh[(mt * 16 + (lane & 15)) * Csz + k * 32 + KA];
  } else if (i < 262144) {                         // W_ph: 16 Mtiles x 16 K
    f = i - 131072; j = f & 15; lane = (f >> 4) & 31; kf = f >> 9;
    kh = lane >> 4; KA = j + 16 * kh;
    int mt = kf >> 4, k = kf & 15;
    wphF[f] = (_Float16)Wph[(mt * 16 + (lane & 15)) * Hsz + k * 32 + KA];
  } else {                                         // W_hx: 32 Mtiles x 4 K
    f = i - 262144; j = f & 15; lane = (f >> 4) & 31; kf = f >> 9;
    kh = lane >> 4; KA = j + 16 * kh;
    int mt = kf >> 2, k = kf & 3;
    whxF[f] = (_Float16)Whx[(mt * 16 + (lane & 15)) * Ssz + k * 32 + KA];
  }
}

// ---------------------------------------------------------------------------
// Prep 2: x (8192x128) f32 -> f16 B-fragments of x^T (S x B):
// layout [ntile 512][kchunk 4][lane 32][j 16]; B in the same k order as A: lane holds column
// N = lane&15; element j -> K = j + 16*(lane>>4) in the 32-wide chunk.
// ---------------------------------------------------------------------------
__global__ __launch_bounds__(256)
void prep_x(const float* __restrict__ x, _Float16* __restrict__ xF) {
  int f = blockIdx.x * blockDim.x + threadIdx.x;   // 0 .. 1048575
  int j    = f & 15;
  int lane = (f >> 4) & 31;
  int rest = f >> 9;
  int kc   = rest & 3;
  int g    = rest >> 2;                            // global N-tile (0..511)
  int b    = g * 16 + (lane & 15);
  int K    = kc * 32 + j + 16 * (lane >> 4);
  xF[f] = (_Float16)x[b * Ssz + K];
}

// ---------------------------------------------------------------------------
// Main: one workgroup owns a 64-wide batch tile for the full 128-step scan.
// LDS: yfrag 32KB | hfrag 64KB | a-tile (f32 D-frag order) 128KB | red 2x2KB.
// ---------------------------------------------------------------------------
__global__ __launch_bounds__(256)
void rnn_main(const float* __restrict__ bh, const float* __restrict__ bp,
              const _Float16* __restrict__ whhF, const _Float16* __restrict__ wphF,
              const _Float16* __restrict__ whxF, const _Float16* __restrict__ xF,
              float* __restrict__ out) {
  extern __shared__ char smem[];
  _Float16* yfrag = (_Float16*)smem;                   // 16384 halves (32 KB)
  _Float16* hfrag = (_Float16*)(smem + 32 * 1024);     // 32768 halves (64 KB)
  float*    aLDS  = (float*)(smem + 96 * 1024);        // 32768 f32   (128 KB)
  float*    red   = (float*)(smem + 224 * 1024);       // [col][wave] 2 KB
  float*    red2  = red + 512;                         // [col][wave] 2 KB

  const int tid   = threadIdx.x;
  const int lane  = tid & 31;
  const int w     = tid >> 5;
  const int n     = lane & 15;
  const int kh    = lane >> 4;
  const int bbase = blockIdx.x * BT;

  // y0 = zeros
  for (int i = tid; i < 8192; i += 256) ((uint32_t*)yfrag)[i] = 0u;

  // b_p values this lane will need (constant over steps)
  float bpv[2][8];
#pragma unroll
  for (int mi = 0; mi < 2; ++mi)
#pragma unroll
    for (int r = 0; r < 8; ++r)
      bpv[mi][r] = bp[(w * 2 + mi) * 16 + r + 8 * kh];

  // ---- Prologue: a = Whx @ x^T (+ b_h) -> aLDS (D-fragment order) ---------
#pragma unroll 1
  for (int mi = 0; mi < 4; ++mi) {
    const int mtg = w * 4 + mi;
    v8f acc[4] = {{}, {}, {}, {}};
#pragma unroll
    for (int kc = 0; kc < 4; ++kc) {
      v16h af = *(const v16h*)&whxF[(size_t)((mtg * 4 + kc) * 32 + lane) * 16];
#pragma unroll
      for (int nt = 0; nt < 4; ++nt) {
        v16h bf = *(const v16h*)&xF[(size_t)(((blockIdx.x * 4 + nt) * 4 + kc) * 32 + lane) * 16];
        acc[nt] = wmma_f16(af, bf, acc[nt]);
      }
    }
#pragma unroll
    for (int nt = 0; nt < 4; ++nt) {
      float bhv = bh[bbase + nt * 16 + n];
#pragma unroll
      for (int r = 0; r < 8; ++r) acc[nt][r] += bhv;
      *(v8f*)&aLDS[((mtg * 4 + nt) * 32 + lane) * 8] = acc[nt];
    }
  }
  __syncthreads();

  // ---- 128-step scan -------------------------------------------------------
#pragma unroll 1
  for (int s = 0; s < Ssz; ++s) {
    // GEMM1: linear = a + Whh @ y^T ; h = tanh(linear) -> hfrag (B-frag f16)
    // Process M-tiles in pairs so each y B-fragment load feeds two WMMAs.
#pragma unroll 1
    for (int mip = 0; mip < 2; ++mip) {
      const int mtg0 = w * 4 + mip * 2;
      const int mtg1 = mtg0 + 1;
      v16h A0[8], A1[8];
#pragma unroll
      for (int k = 0; k < 8; ++k) {
        A0[k] = *(const v16h*)&whhF[(size_t)((mtg0 * 8 + k) * 32 + lane) * 16];
        A1[k] = *(const v16h*)&whhF[(size_t)((mtg1 * 8 + k) * 32 + lane) * 16];
      }
#pragma unroll
      for (int nt = 0; nt < 4; ++nt) {
        v8f acc0 = *(const v8f*)&aLDS[((mtg0 * 4 + nt) * 32 + lane) * 8];
        v8f acc1 = *(const v8f*)&aLDS[((mtg1 * 4 + nt) * 32 + lane) * 8];
#pragma unroll
        for (int k = 0; k < 8; ++k) {
          v16h bf = *(const v16h*)&yfrag[((nt * 8 + k) * 32 + lane) * 16];
          acc0 = wmma_f16(A0[k], bf, acc0);
          acc1 = wmma_f16(A1[k], bf, acc1);
        }
#pragma unroll
        for (int r = 0; r < 8; r += 2) {
          int m0 = mtg0 * 16 + r + 8 * kh;       // H row (even)
          int k2 = m0 >> 5, kk = m0 & 31;
          int lane2 = n + 16 * ((kk >> 4) & 1);
          h2v hv0 = {(_Float16)fast_tanh(acc0[r]), (_Float16)fast_tanh(acc0[r + 1])};
          *(h2v*)&hfrag[((nt * 16 + k2) * 32 + lane2) * 16 + (kk & 15)] = hv0;

          int m1 = mtg1 * 16 + r + 8 * kh;
          int k2b = m1 >> 5, kkb = m1 & 31;
          int lane2b = n + 16 * ((kkb >> 4) & 1);
          h2v hv1 = {(_Float16)fast_tanh(acc1[r]), (_Float16)fast_tanh(acc1[r + 1])};
          *(h2v*)&hfrag[((nt * 16 + k2b) * 32 + lane2b) * 16 + (kkb & 15)] = hv1;
        }
      }
    }
    __syncthreads();

    // GEMM2: p = Wph @ h + b_p, kept in registers (64 f32/lane).
    // k blocked 4x4 to bound live operand staging on top of 64 acc VGPRs.
    v8f acc2[2][4] = {{{}, {}, {}, {}}, {{}, {}, {}, {}}};
#pragma unroll 1
    for (int kb = 0; kb < 4; ++kb) {
#pragma unroll
      for (int ki = 0; ki < 4; ++ki) {
        const int k = kb * 4 + ki;
        v16h af0 = *(const v16h*)&wphF[(size_t)(((w * 2) * 16 + k) * 32 + lane) * 16];
        v16h af1 = *(const v16h*)&wphF[(size_t)(((w * 2 + 1) * 16 + k) * 32 + lane) * 16];
#pragma unroll
        for (int nt = 0; nt < 4; ++nt) {
          v16h bf = *(const v16h*)&hfrag[((nt * 16 + k) * 32 + lane) * 16];
          acc2[0][nt] = wmma_f16(af0, bf, acc2[0][nt]);
          acc2[1][nt] = wmma_f16(af1, bf, acc2[1][nt]);
        }
      }
    }

    // + b_p, per-column partial max over this wave's 32 rows
    float pmx[4] = {-3.4e38f, -3.4e38f, -3.4e38f, -3.4e38f};
#pragma unroll
    for (int mi = 0; mi < 2; ++mi)
#pragma unroll
      for (int nt = 0; nt < 4; ++nt)
#pragma unroll
        for (int r = 0; r < 8; ++r) {
          float v = acc2[mi][nt][r] + bpv[mi][r];
          acc2[mi][nt][r] = v;
          pmx[nt] = fmaxf(pmx[nt], v);
        }
#pragma unroll
    for (int nt = 0; nt < 4; ++nt) {
      pmx[nt] = fmaxf(pmx[nt], __shfl_xor(pmx[nt], 16, 32));
      red[(nt * 16 + n) * 8 + w] = pmx[nt];      // both kh halves: same value
    }
    __syncthreads();

    // global max per column (one v8f LDS read), exp, partial sums
    float cmax[4];
#pragma unroll
    for (int nt = 0; nt < 4; ++nt) {
      v8f rv = *(const v8f*)&red[(nt * 16 + n) * 8];
      float m = rv[0];
#pragma unroll
      for (int w8 = 1; w8 < 8; ++w8) m = fmaxf(m, rv[w8]);
      cmax[nt] = m;
    }
    float psm[4] = {0.f, 0.f, 0.f, 0.f};
#pragma unroll
    for (int mi = 0; mi < 2; ++mi)
#pragma unroll
      for (int nt = 0; nt < 4; ++nt)
#pragma unroll
        for (int r = 0; r < 8; ++r) {
          float e = __expf(acc2[mi][nt][r] - cmax[nt]);
          acc2[mi][nt][r] = e;
          psm[nt] += e;
        }
#pragma unroll
    for (int nt = 0; nt < 4; ++nt) {
      psm[nt] += __shfl_xor(psm[nt], 16, 32);
      red2[(nt * 16 + n) * 8 + w] = psm[nt];
    }
    __syncthreads();

    // normalize, write y into yfrag (B-frag f16); last step -> global out
#pragma unroll
    for (int nt = 0; nt < 4; ++nt) {
      v8f sv = *(const v8f*)&red2[(nt * 16 + n) * 8];
      float ssum = 0.f;
#pragma unroll
      for (int w8 = 0; w8 < 8; ++w8) ssum += sv[w8];
      float inv = 1.0f / ssum;
      const int col = nt * 16 + n;
#pragma unroll
      for (int mi = 0; mi < 2; ++mi)
#pragma unroll
        for (int r = 0; r < 8; r += 2) {
          int cc = (w * 2 + mi) * 16 + r + 8 * kh;   // even
          float y0 = acc2[mi][nt][r] * inv;
          float y1 = acc2[mi][nt][r + 1] * inv;
          int kc = cc >> 5, kk = cc & 31;
          int lane2 = n + 16 * ((kk >> 4) & 1);
          h2v hv = {(_Float16)y0, (_Float16)y1};
          *(h2v*)&yfrag[((nt * 8 + kc) * 32 + lane2) * 16 + (kk & 15)] = hv;
          if (s == Ssz - 1) {
            f2v o = {y0, y1};
            *(f2v*)&out[(size_t)(bbase + col) * Csz + cc] = o;
          }
        }
    }
    __syncthreads();
  }
}

// ---------------------------------------------------------------------------
extern "C" void kernel_launch(void* const* d_in, const int* in_sizes, int n_in,
                              void* d_out, int out_size, void* d_ws, size_t ws_size,
                              hipStream_t stream) {
  const float* x   = (const float*)d_in[0];
  const float* Whx = (const float*)d_in[1];
  const float* Whh = (const float*)d_in[2];
  const float* bh  = (const float*)d_in[3];
  const float* Wph = (const float*)d_in[4];
  const float* bp  = (const float*)d_in[5];

  _Float16* whhF = (_Float16*)d_ws;          // 131072 halves (256 KB)
  _Float16* wphF = whhF + 131072;            // 131072 halves (256 KB)
  _Float16* whxF = wphF + 131072;            //  65536 halves (128 KB)
  _Float16* xF   = whxF + 65536;             // 1048576 halves (2 MB)

  prep_weights<<<1280, 256, 0, stream>>>(Whh, Wph, Whx, whhF, wphF, whxF);
  prep_x<<<4096, 256, 0, stream>>>(x, xF);

  const size_t lds = 224 * 1024 + 4096;      // 233472 bytes
  rnn_main<<<Bsz / BT, 256, lds, stream>>>(bh, bp, whhF, wphF, whxF, xF,
                                           (float*)d_out);
}
